// STConvBlock_10050223473195
// MI455X (gfx1250) — matrix operand order repaired and checked by reading, not compiled, not run on hardware
//
#include <hip/hip_runtime.h>

#define Bc 8
#define Tc 12
#define Nn 400
#define C1c 64
#define C2c 128
#define T1 10
#define T2 8
#define BT1 (Bc*T1)
#define NT 25
#define KP 208   // padded j-pairs (416/2)
#define XFP_W (KP*64)   // 13312 dwords per (b,t) slab

typedef __attribute__((ext_vector_type(16))) __bf16 bf16x16;
typedef __attribute__((ext_vector_type(8)))  float  f32x8;
typedef __attribute__((ext_vector_type(4)))  unsigned u32x4;
typedef __attribute__((ext_vector_type(8)))  int      i32x8;
typedef __attribute__((ext_vector_type(4)))  int      i32x4;

union BFrag { unsigned u[8]; bf16x16 v; };

__device__ __forceinline__ unsigned short f2bf(float f) {
  unsigned u = __float_as_uint(f);
  u += 0x7FFFu + ((u >> 16) & 1u);   // round-to-nearest-even
  return (unsigned short)(u >> 16);
}
__device__ __forceinline__ unsigned pack2(float a, float b) {
  return (unsigned)f2bf(a) | ((unsigned)f2bf(b) << 16);
}
__device__ __forceinline__ float bf_lo(unsigned v) { return __uint_as_float(v << 16); }
__device__ __forceinline__ float bf_hi(unsigned v) { return __uint_as_float(v & 0xFFFF0000u); }
__device__ __forceinline__ f32x8 wmma_bf16(const BFrag& A, const BFrag& B, f32x8 C) {
  return __builtin_amdgcn_wmma_f32_16x16x32_bf16(false, A.v, false, B.v, (short)0, C, false, false);
}

// TDM: copy `bytes` (multiple of 8) from global -> start of dynamic LDS.
__device__ __forceinline__ void tdm_load_to_lds(const void* gptr, unsigned bytes) {
  unsigned long long ga = (unsigned long long)gptr;
  unsigned elems = bytes >> 3;                       // data_size = 8B
  u32x4 g0;
  g0.x = 1u;                                         // count=1, user mode
  g0.y = __builtin_amdgcn_groupstaticsize();         // lds_addr = dynamic LDS base
  g0.z = (unsigned)(ga & 0xFFFFFFFFu);               // global_addr[31:0]
  g0.w = (unsigned)((ga >> 32) & 0x1FFFFFFu) | (2u << 30);  // addr[56:32] | type=2
  i32x8 g1;
  g1[0] = (int)(3u << 16);                           // data_size code 3 (8B)
  g1[1] = (int)((elems & 0xFFFFu) << 16);            // tensor_dim0[15:0]
  g1[2] = (int)((elems >> 16) | (1u << 16));         // tensor_dim0[31:16] | tensor_dim1=1
  g1[3] = (int)((elems & 0xFFFFu) << 16);            // tile_dim0
  g1[4] = 0;                                         // tile_dim1=0 (1-D tile)
  g1[5] = (int)elems;                                // tensor_dim0_stride[31:0]
  g1[6] = 0; g1[7] = 0;
  i32x4 gz = {0, 0, 0, 0};
#if __clang_major__ >= 23
  i32x8 gz8 = {0,0,0,0,0,0,0,0};
  __builtin_amdgcn_tensor_load_to_lds(g0, g1, gz, gz, gz8, 0);
#else
  __builtin_amdgcn_tensor_load_to_lds(g0, g1, gz, gz, 0);
#endif
}

// ---------------- K0: pack bf16 pair layouts + u vectors ----------------
__global__ void k0_prep(const float* __restrict__ inp, const float* __restrict__ wt1,
                        const float* __restrict__ wt2, const float* __restrict__ wcheb,
                        const float* __restrict__ wf, const float* __restrict__ asrc,
                        const float* __restrict__ adst, const float* __restrict__ sup,
                        unsigned* inb, unsigned* wt1p, unsigned* wt2p, unsigned* wchp,
                        unsigned* supb, float* usrc, float* udst) {
  const int N0 = Bc*Tc*Nn*32;          // input c-pairs
  const int N1 = 96*128, N3 = 4*4*32*64, N4 = 512, N5 = 4*Nn*KP;
  const int total = N0 + N1 + N1 + N3 + N4 + N5;
  for (int idx = blockIdx.x*blockDim.x + threadIdx.x; idx < total;
       idx += gridDim.x*blockDim.x) {
    int t = idx;
    if (t < N0) { inb[t] = pack2(inp[2*t], inp[2*t+1]); continue; }
    t -= N0;
    if (t < N1) {
      int kp = t >> 7, o = t & 127, kt = kp >> 5, cp = kp & 31;
      int base = (kt*64 + 2*cp)*128 + o;
      wt1p[t] = pack2(wt1[base], wt1[base+128]); continue;
    }
    t -= N1;
    if (t < N1) {
      int kp = t >> 7, o = t & 127, kt = kp >> 5, cp = kp & 31;
      int base = (kt*64 + 2*cp)*128 + o;
      wt2p[t] = pack2(wt2[base], wt2[base+128]); continue;
    }
    t -= N1;
    if (t < N3) {
      int o = t & 63, cp = (t>>6)&31, hk = t>>11;   // hk = h*4+k
      int base = (hk*64 + 2*cp)*64 + o;
      wchp[t] = pack2(wcheb[base], wcheb[base+64]); continue;
    }
    t -= N3;
    if (t < N4) {
      int sel = t >> 8, h = (t>>6)&3, c = t&63;
      const float* a = sel ? adst : asrc;
      float s = 0.f;
      for (int o = 0; o < 64; o++) s += wf[(h*64+c)*64+o]*a[h*64+o];
      (sel ? udst : usrc)[h*64+c] = s;
      continue;
    }
    t -= N4;
    {   // supports -> bf16 j-pairs: supb[k][i][jp]
      int k = t / (Nn*KP), rem = t % (Nn*KP);
      int i = rem / KP, jp = rem % KP, j0 = 2*jp;
      unsigned v = 0;
      if (j0 < Nn) {
        const float* Sp = sup + ((size_t)k*Nn + i)*Nn + j0;
        v = pack2(Sp[0], Sp[1]);
      }
      supb[t] = v;
    }
  }
}

// ---------------- K1: conv1 (WMMA) + GLU -> x ----------------
__global__ void k1_conv1_glu(const unsigned* __restrict__ inb, const unsigned* __restrict__ wt1p,
                             const float* __restrict__ bt1, const float* __restrict__ inp,
                             float* __restrict__ x) {
  int b = blockIdx.x / T1, t = blockIdx.x % T1;
  int tid = threadIdx.x, w = tid>>5, L = tid&31, m = L&15, hi = L>>4;
  float b1a = bt1[w*16 + m], b1b = bt1[64 + w*16 + m];
  for (int mt = w; mt < NT; mt += 8) {
    f32x8 accA = {0,0,0,0,0,0,0,0}, accB = {0,0,0,0,0,0,0,0};
    for (int ks = 0; ks < 6; ks++) {
      int kt = ks>>1, half = ks&1;
      BFrag A, B0, B1;
      const unsigned* ar = inb + (((b*Tc + t + kt)*Nn) + mt*16 + m)*32 + half*16;
      #pragma unroll
      for (int r = 0; r < 8; r++) {
        int kpl = (hi<<3) + r;
        A.u[r] = ar[kpl];
      }
      #pragma unroll
      for (int r = 0; r < 8; r++) {
        int pr = ks*16 + hi*8 + r;
        B0.u[r] = wt1p[pr*128 + w*16 + m];
        B1.u[r] = wt1p[pr*128 + 64 + w*16 + m];
      }
      accA = wmma_bf16(A, B0, accA);
      accB = wmma_bf16(A, B1, accB);
    }
    #pragma unroll
    for (int r = 0; r < 8; r++) {
      int n = mt*16 + r + 8*hi, c = w*16 + m;
      float res = inp[(((b*Tc + t + 2)*Nn) + n)*64 + c];
      float ca = accA[r] + b1a, cb = accB[r] + b1b;
      float sg = __builtin_amdgcn_rcpf(1.0f + __expf(-cb));
      x[((b*T1 + t)*Nn + n)*64 + c] = (ca + res) * sg;
    }
  }
}

// ---------------- K2: e_src/e_dst via folded Wf·a ----------------
__global__ void k2_e(const float* __restrict__ x, const float* __restrict__ usrc,
                     const float* __restrict__ udst, float* __restrict__ esrc,
                     float* __restrict__ edst) {
  __shared__ float us[256], ud[256];
  int tid = threadIdx.x;
  us[tid] = usrc[tid]; ud[tid] = udst[tid];
  __syncthreads();
  for (int idx = blockIdx.x*blockDim.x + tid; idx < BT1*Nn; idx += gridDim.x*blockDim.x) {
    float es[4] = {0,0,0,0}, ed[4] = {0,0,0,0};
    const float* xr = x + (size_t)idx*64;
    for (int c = 0; c < 64; c++) {
      float xv = xr[c];
      #pragma unroll
      for (int h = 0; h < 4; h++) { es[h] += xv*us[h*64+c]; ed[h] += xv*ud[h*64+c]; }
    }
    int bt = idx / Nn, n = idx % Nn;
    #pragma unroll
    for (int h = 0; h < 4; h++) {
      esrc[(bt*4+h)*Nn + n] = es[h];
      edst[(bt*4+h)*Nn + n] = ed[h];
    }
  }
}

// ---------------- K2b: pack xf -> bf16 j-pair slabs (TDM source) ----------------
__global__ void k2b_pack_xf(const float* __restrict__ x, unsigned* __restrict__ xfp) {
  const int total = BT1*XFP_W;
  for (int idx = blockIdx.x*blockDim.x + threadIdx.x; idx < total;
       idx += gridDim.x*blockDim.x) {
    int bt = idx / XFP_W, r = idx % XFP_W;
    int j2 = r >> 6, c = r & 63;
    unsigned v = 0;
    if (j2 < 200) {
      const float* p = x + ((size_t)bt*Nn + 2*j2)*64 + c;
      v = pack2(p[0], p[64]);
    }
    xfp[idx] = v;
  }
}

// ---------------- K3: attention softmax + cheb message passing (WMMA + TDM) ----------------
__global__ void k3_attn(const float* __restrict__ x, const unsigned* __restrict__ xfp,
                        const unsigned* __restrict__ supb,
                        const unsigned char* __restrict__ adj,
                        const float* __restrict__ esrc_g, const float* __restrict__ edst_g,
                        const unsigned* __restrict__ wchp,
                        float* __restrict__ x2, unsigned short* __restrict__ x2b) {
  extern __shared__ unsigned char smem[];
  unsigned* xf_p  = (unsigned*)smem;            // [208][64] bf16 j-pairs of xf (TDM target)
  unsigned* A_p   = xf_p + XFP_W;               // [4][16][208] bf16 pairs of alpha*S_k
  float*    alpha = (float*)(A_p + 4*16*KP);    // [16][400] (reused as msg_s [4][16][64])
  float*    edst_s = alpha + 6400;              // [400]
  float*    inv_r  = edst_s + 400;              // [16]
  float*    msg_s  = alpha;

  int bt = blockIdx.x / NT, itile = blockIdx.x % NT;
  int tid = threadIdx.x, w = tid>>5, L = tid&31, m = L&15, hi = L>>4;

  // stage xf[bt] slab into LDS via Tensor Data Mover (wave 0 issues, all wait)
  if (w == 0) {
    tdm_load_to_lds(xfp + (size_t)bt*XFP_W, XFP_W*4u);
    __builtin_amdgcn_s_wait_tensorcnt(0);
  }

  f32x8 out_acc = {0,0,0,0,0,0,0,0};
  for (int h = 0; h < 4; h++) {
    __syncthreads();
    for (int j = tid; j < Nn; j += 128) edst_s[j] = edst_g[(bt*4+h)*Nn + j];
    __syncthreads();

    // softmax: 8 threads per row, 16 rows
    int i = tid >> 3, lr = tid & 7;
    int ig = itile*16 + i;
    float es = esrc_g[(bt*4+h)*Nn + ig];
    float mx = -3.0e38f;
    for (int j = lr; j < Nn; j += 8) {
      float l = es + edst_s[j];
      l = l > 0.f ? l : 0.2f*l;
      l = adj[ig*Nn + j] ? l : -1.0e9f;
      alpha[i*Nn + j] = l;
      mx = fmaxf(mx, l);
    }
    mx = fmaxf(mx, __shfl_xor(mx, 1, 32));
    mx = fmaxf(mx, __shfl_xor(mx, 2, 32));
    mx = fmaxf(mx, __shfl_xor(mx, 4, 32));
    float s = 0.f;
    for (int j = lr; j < Nn; j += 8) {
      float e = __expf(alpha[i*Nn + j] - mx);
      alpha[i*Nn + j] = e;
      s += e;
    }
    s += __shfl_xor(s, 1, 32);
    s += __shfl_xor(s, 2, 32);
    s += __shfl_xor(s, 4, 32);
    if (lr == 0) inv_r[i] = __builtin_amdgcn_rcpf(s);
    __syncthreads();

    // build A_k = (alpha/sum) * S_k as bf16 pairs (S pre-packed bf16)
    for (int idx = tid; idx < 4*16*KP; idx += 128) {
      int k = idx / (16*KP), rem = idx % (16*KP);
      int ii = rem / KP, jp = rem % KP, j0 = jp*2;
      unsigned v = 0;
      if (j0 < Nn) {
        int irow = itile*16 + ii;
        float inv = inv_r[ii];
        unsigned sv = supb[((size_t)k*Nn + irow)*KP + jp];
        v = pack2(alpha[ii*Nn + j0] * inv * bf_lo(sv),
                  alpha[ii*Nn + j0 + 1] * inv * bf_hi(sv));
      }
      A_p[idx] = v;
    }
    __syncthreads();

    // msg GEMM: [16x416] x [416x64] per k; wave w owns c-tile w
    f32x8 acc[4];
    #pragma unroll
    for (int k = 0; k < 4; k++) { f32x8 z = {0,0,0,0,0,0,0,0}; acc[k] = z; }
    for (int ks = 0; ks < 13; ks++) {
      BFrag Bt;
      #pragma unroll
      for (int r = 0; r < 8; r++) Bt.u[r] = xf_p[(ks*16 + hi*8 + r)*64 + w*16 + m];
      #pragma unroll
      for (int k = 0; k < 4; k++) {
        BFrag At;
        #pragma unroll
        for (int r = 0; r < 8; r++) {
          int kpl = (hi<<3) + r;
          At.u[r] = A_p[(k*16 + m)*KP + ks*16 + kpl];
        }
        acc[k] = wmma_bf16(At, Bt, acc[k]);
      }
    }
    // dump msg to LDS (f32) for the Wcheb GEMM
    #pragma unroll
    for (int k = 0; k < 4; k++)
      #pragma unroll
      for (int r = 0; r < 8; r++)
        msg_s[(k*16 + r + 8*hi)*64 + w*16 + m] = acc[k][r];
    __syncthreads();

    // out += sum_k msg_k @ Wcheb[h,k]  (K=64 -> 2 ksteps per k)
    #pragma unroll
    for (int k = 0; k < 4; k++) {
      #pragma unroll
      for (int ks2 = 0; ks2 < 2; ks2++) {
        BFrag At, Bt;
        #pragma unroll
        for (int r = 0; r < 8; r++) {
          int c = ks2*32 + (hi<<4) + (r<<1);
          const float* pm = &msg_s[(k*16 + m)*64 + c];
          At.u[r] = pack2(pm[0], pm[1]);
          int pr = ks2*16 + hi*8 + r;
          Bt.u[r] = wchp[((h*4 + k)*32 + pr)*64 + w*16 + m];
        }
        out_acc = wmma_bf16(At, Bt, out_acc);
      }
    }
    __syncthreads();
  }

  // x2 = relu(gc/H + x), fp32 + bf16 copy for conv2
  #pragma unroll
  for (int r = 0; r < 8; r++) {
    int n = itile*16 + r + 8*hi, c = w*16 + m;
    size_t gi = ((size_t)bt*Nn + n)*64 + c;
    float v = out_acc[r]*0.25f + x[gi];
    v = v > 0.f ? v : 0.f;
    x2[gi] = v;
    x2b[gi] = f2bf(v);
  }
}

// ---------------- K4: conv2 (WMMA) + res2 + relu -> y ----------------
__global__ void k4_conv2(const unsigned short* __restrict__ x2b_, const unsigned* __restrict__ wt2p,
                         const float* __restrict__ bt2, const float* __restrict__ x2,
                         float* __restrict__ y) {
  const unsigned* x2b = (const unsigned*)x2b_;
  int b = blockIdx.x / T2, t = blockIdx.x % T2;
  int tid = threadIdx.x, w = tid>>5, L = tid&31, m = L&15, hi = L>>4;
  unsigned breg[6][8];
  #pragma unroll
  for (int ks = 0; ks < 6; ks++)
    #pragma unroll
    for (int r = 0; r < 8; r++)
      breg[ks][r] = wt2p[(ks*16 + hi*8 + r)*128 + w*16 + m];
  float bias = bt2[w*16 + m];
  for (int mt = 0; mt < NT; mt++) {
    f32x8 acc = {0,0,0,0,0,0,0,0};
    #pragma unroll
    for (int ks = 0; ks < 6; ks++) {
      int kt = ks>>1, half = ks&1;
      BFrag A, Bt;
      const unsigned* ar = x2b + (((size_t)(b*T1) + t + kt)*Nn + mt*16 + m)*32 + half*16;
      #pragma unroll
      for (int r = 0; r < 8; r++) {
        int kpl = (hi<<3) + r;
        A.u[r] = ar[kpl];
      }
      #pragma unroll
      for (int r = 0; r < 8; r++) Bt.u[r] = breg[ks][r];
      acc = wmma_bf16(A, Bt, acc);
    }
    #pragma unroll
    for (int r = 0; r < 8; r++) {
      int n = mt*16 + r + 8*hi, o = w*16 + m;
      float res = (o < 64) ? x2[((size_t)(b*T1 + t + 2)*Nn + n)*64 + o] : 0.f;
      float v = acc[r] + bias + res;
      y[((size_t)(b*T2 + t)*Nn + n)*128 + o] = v > 0.f ? v : 0.f;
    }
  }
}

// ---------------- K5: per-(b,t) mean / inv-std ----------------
__global__ void k5_stats(const float* __restrict__ y, float* __restrict__ stats) {
  int bt = blockIdx.x, tid = threadIdx.x;
  const float* p = y + (size_t)bt*51200;
  float s = 0.f, q = 0.f;
  for (int i = tid; i < 51200; i += 256) { float v = p[i]; s += v; q += v*v; }
  for (int o = 1; o < 32; o <<= 1) { s += __shfl_xor(s, o, 32); q += __shfl_xor(q, o, 32); }
  __shared__ float ss[8], qq[8];
  if ((tid & 31) == 0) { ss[tid>>5] = s; qq[tid>>5] = q; }
  __syncthreads();
  if (tid == 0) {
    float S = 0.f, Q = 0.f;
    for (int i = 0; i < 8; i++) { S += ss[i]; Q += qq[i]; }
    float mu = S / 51200.0f;
    float var = Q / 51200.0f - mu*mu;
    stats[bt*2] = mu;
    stats[bt*2+1] = rsqrtf(var + 1e-6f);
  }
}

// ---------------- K6: normalize + affine ----------------
__global__ void k6_norm(const float* __restrict__ y, const float* __restrict__ stats,
                        const float* __restrict__ gamma, const float* __restrict__ beta,
                        float* __restrict__ out) {
  const int total = Bc*T2*Nn*C2c;
  for (int idx = blockIdx.x*blockDim.x + threadIdx.x; idx < total; idx += gridDim.x*blockDim.x) {
    int bt = idx / 51200, nc = idx % 51200;
    out[idx] = (y[idx] - stats[bt*2]) * stats[bt*2+1] * gamma[nc] + beta[nc];
  }
}

extern "C" void kernel_launch(void* const* d_in, const int* in_sizes, int n_in,
                              void* d_out, int out_size, void* d_ws, size_t ws_size,
                              hipStream_t stream) {
  (void)in_sizes; (void)n_in; (void)out_size; (void)ws_size;
  const float* inp  = (const float*)d_in[0];
  const float* sup  = (const float*)d_in[1];
  const unsigned char* adj = (const unsigned char*)d_in[2];
  const float* wt1  = (const float*)d_in[3];
  const float* bt1  = (const float*)d_in[4];
  const float* wf   = (const float*)d_in[5];
  const float* asrc = (const float*)d_in[6];
  const float* adst = (const float*)d_in[7];
  const float* wch  = (const float*)d_in[8];
  const float* wt2  = (const float*)d_in[9];
  const float* bt2  = (const float*)d_in[10];
  const float* gma  = (const float*)d_in[11];
  const float* bta  = (const float*)d_in[12];
  float* out = (float*)d_out;

  unsigned char* ws = (unsigned char*)d_ws;
  size_t off = 0;
  auto alloc = [&](size_t bytes) -> void* {
    void* p = ws + off;
    off = (off + bytes + 255) & ~(size_t)255;
    return p;
  };
  float*    x    = (float*)alloc((size_t)BT1*Nn*64*4);
  float*    x2   = (float*)alloc((size_t)BT1*Nn*64*4);
  float*    yb   = (float*)alloc((size_t)Bc*T2*Nn*128*4);
  unsigned* inb  = (unsigned*)alloc((size_t)Bc*Tc*Nn*32*4);
  unsigned* wt1p = (unsigned*)alloc(96*128*4);
  unsigned* wt2p = (unsigned*)alloc(96*128*4);
  unsigned* wchp = (unsigned*)alloc(4*4*32*64*4);
  unsigned* supb = (unsigned*)alloc((size_t)4*Nn*KP*4);
  unsigned* xfp  = (unsigned*)alloc((size_t)BT1*XFP_W*4);
  float*    usrc = (float*)alloc(256*4);
  float*    udst = (float*)alloc(256*4);
  float*    esrc = (float*)alloc((size_t)BT1*4*Nn*4);
  float*    edst = (float*)alloc((size_t)BT1*4*Nn*4);
  unsigned short* x2b = (unsigned short*)alloc((size_t)BT1*Nn*64*2);
  float*    stats = (float*)alloc(128*4);

  k0_prep<<<2048, 256, 0, stream>>>(inp, wt1, wt2, wch, wf, asrc, adst, sup,
                                    inb, wt1p, wt2p, wchp, supb, usrc, udst);
  k1_conv1_glu<<<Bc*T1, 256, 0, stream>>>(inb, wt1p, bt1, inp, x);
  k2_e<<<125, 256, 0, stream>>>(x, usrc, udst, esrc, edst);
  k2b_pack_xf<<<2080, 256, 0, stream>>>(x, xfp);
  size_t shmem = (size_t)(XFP_W + 4*16*KP)*4 + (size_t)(6400 + 400 + 16)*4; // ~134 KB
  k3_attn<<<BT1*NT, 128, shmem, stream>>>(x, xfp, supb, adj, esrc, edst, wchp, x2, x2b);
  k4_conv2<<<Bc*T2, 256, 0, stream>>>(x2b, wt2p, bt2, x2, yb);
  k5_stats<<<Bc*T2, 256, 0, stream>>>(yb, stats);
  k6_norm<<<12800, 256, 0, stream>>>(yb, stats, gma, bta, out);
}
